// GPRGNN_26645977105009
// MI455X (gfx1250) — matrix operand order repaired and checked by reading, not compiled, not run on hardware
//
#include <hip/hip_runtime.h>

#define N_NODES 100000
#define N_EDGES 3200000
#define NFEAT   512
#define NHID    64
#define NCLASS  64
#define KPROP   10

typedef __attribute__((ext_vector_type(16))) _Float16 v16h;
typedef __attribute__((ext_vector_type(8)))  float    v8f;

// ---------------- gcn_norm ----------------

__global__ void zero_deg_kernel(int* deg) {
    int i = blockIdx.x * blockDim.x + threadIdx.x;
    if (i < N_NODES) deg[i] = 0;
}

__global__ void deg_kernel(const int* __restrict__ dst, int* deg) {
    int e = blockIdx.x * blockDim.x + threadIdx.x;
    if (e < N_EDGES) atomicAdd(&deg[dst[e]], 1);
}

__global__ void dinv_kernel(const int* __restrict__ deg, float* __restrict__ dinv) {
    int i = blockIdx.x * blockDim.x + threadIdx.x;
    if (i < N_NODES) {
        float d = (float)deg[i] + 1.0f;   // +1 self loop; always > 0
        dinv[i] = rsqrtf(d);
    }
}

// ------------- weight pre-transpose (f32 -> f16, [K][N] -> [N][K]) -------------

__global__ void prep_weights_kernel(const float* __restrict__ W1, const float* __restrict__ W2,
                                    _Float16* __restrict__ w1t, _Float16* __restrict__ w2t) {
    int idx = blockIdx.x * blockDim.x + threadIdx.x;
    if (idx < NFEAT * NHID) {
        int k = idx / NHID, n = idx % NHID;
        w1t[n * NFEAT + k] = (_Float16)W1[idx];
    }
    if (idx < NHID * NCLASS) {
        int k = idx / NCLASS, n = idx % NCLASS;
        w2t[n * NHID + k] = (_Float16)W2[idx];
    }
}

// ------------- MLP: h = relu(x@W1+b1)@W2 + b2 ; also hidden = temp[0]*h -------------
// 256 threads = 8 waves; each wave computes a 16x64 output stripe with
// v_wmma_f32_16x16x32_f16 (4 accumulators of 16x16).

__global__ __launch_bounds__(256) void mlp_wmma_kernel(
    const float* __restrict__ x,
    const _Float16* __restrict__ w1t,   // [NHID][NFEAT] f16
    const float* __restrict__ b1,
    const _Float16* __restrict__ w2t,   // [NCLASS][NHID] f16
    const float* __restrict__ b2,
    const float* __restrict__ temp,
    float* __restrict__ h0,
    float* __restrict__ out) {

    // padded stride 68 -> bank = (4*m + k) % 64, conflict-free column reads
    __shared__ float zbuf[8][16][NHID + 4];

    const int tid  = threadIdx.x;
    const int wave = tid >> 5;
    const int lane = tid & 31;
    const int hsel = (lane >> 4) & 1;   // which half of the wave
    const int lm   = lane & 15;
    const int r0   = blockIdx.x * 128 + wave * 16;

    // ---- Layer 1: [16 x 512] @ [512 x 64] ----
    int rowA = r0 + lm;
    if (rowA > N_NODES - 1) rowA = N_NODES - 1;   // clamp loads; stores guarded later
    const float* xrow = x + (size_t)rowA * NFEAT;

    v8f acc[4] = {};

    for (int ks = 0; ks < NFEAT / 32; ++ks) {
        const int kb = ks * 32;
        // A fragment (contiguous K, matching B):
        // lanes 0-15: elems 0-15 -> K=kb+0..15
        // lanes 16-31: elems 0-15 -> K=kb+16..31
        v16h a;
        const int k0 = kb + hsel * 16;
#pragma unroll
        for (int i = 0; i < 8; ++i) a[i]     = (_Float16)xrow[k0 + i];
#pragma unroll
        for (int i = 0; i < 8; ++i) a[8 + i] = (_Float16)xrow[k0 + 8 + i];

        const int kbase = kb + hsel * 16;   // B: lanes 0-15 K=kb..kb+15, lanes 16-31 K=kb+16..kb+31
#pragma unroll
        for (int nt = 0; nt < 4; ++nt) {
            const int n = nt * 16 + lm;
            v16h b = *(const v16h*)(w1t + (size_t)n * NFEAT + kbase);  // 32B contiguous
            acc[nt] = __builtin_amdgcn_wmma_f32_16x16x32_f16(
                false, a, false, b, (short)0, acc[nt], false, false);
        }
    }

    // bias + relu -> LDS (re-layout C(M,N) -> A(M,K) for layer 2)
#pragma unroll
    for (int nt = 0; nt < 4; ++nt) {
#pragma unroll
        for (int j = 0; j < 8; ++j) {
            int m   = j + hsel * 8;
            int col = nt * 16 + lm;
            float v = acc[nt][j] + b1[col];
            zbuf[wave][m][col] = v > 0.0f ? v : 0.0f;
        }
    }
    __syncthreads();

    // ---- Layer 2: [16 x 64] @ [64 x 64] ----
    v8f acc2[4] = {};
#pragma unroll
    for (int ks = 0; ks < NHID / 32; ++ks) {
        const int kb = ks * 32;
        v16h a;
        const int k0 = kb + hsel * 16;
#pragma unroll
        for (int i = 0; i < 8; ++i) a[i]     = (_Float16)zbuf[wave][lm][k0 + i];
#pragma unroll
        for (int i = 0; i < 8; ++i) a[8 + i] = (_Float16)zbuf[wave][lm][k0 + 8 + i];

        const int kbase = kb + hsel * 16;
#pragma unroll
        for (int nt = 0; nt < 4; ++nt) {
            const int n = nt * 16 + lm;
            v16h b = *(const v16h*)(w2t + (size_t)n * NHID + kbase);
            acc2[nt] = __builtin_amdgcn_wmma_f32_16x16x32_f16(
                false, a, false, b, (short)0, acc2[nt], false, false);
        }
    }

    const float t0 = temp[0];
#pragma unroll
    for (int nt = 0; nt < 4; ++nt) {
#pragma unroll
        for (int j = 0; j < 8; ++j) {
            int m   = j + hsel * 8;
            int row = r0 + m;
            int col = nt * 16 + lm;
            if (row < N_NODES) {
                float v = acc2[nt][j] + b2[col];
                size_t o = (size_t)row * NCLASS + col;
                h0[o]  = v;
                out[o] = t0 * v;
            }
        }
    }
}

// ------------- propagation: h_next = A_hat @ h_cur -------------

// self-loop term: h_next = dinv[i]^2 * h_cur  (float4 per thread)
__global__ void init_next_kernel(const float* __restrict__ hcur,
                                 const float* __restrict__ dinv,
                                 float* __restrict__ hnext) {
    int idx = blockIdx.x * blockDim.x + threadIdx.x;   // over N*16 float4's
    if (idx < N_NODES * 16) {
        int node = idx >> 4;
        float di = dinv[node];
        float w  = di * di;
        float4 v = ((const float4*)hcur)[idx];
        float4 r = make_float4(w * v.x, w * v.y, w * v.z, w * v.w);
        ((float4*)hnext)[idx] = r;
    }
}

// 16 lanes per edge; each lane handles one float4 of the 64-wide row.
__global__ void edge_kernel(const int* __restrict__ src, const int* __restrict__ dst,
                            const float* __restrict__ dinv,
                            const float* __restrict__ hcur,
                            float* __restrict__ hnext) {
    long long t = (long long)blockIdx.x * blockDim.x + threadIdx.x;
    long long e = t >> 4;
    int sub = (int)(t & 15);
    if (e < N_EDGES) {
        int s = src[e], d = dst[e];
        float w = dinv[s] * dinv[d];
        float4 v = ((const float4*)(hcur + (size_t)s * NCLASS))[sub];
        float* hd = hnext + (size_t)d * NCLASS + sub * 4;
        atomicAdd(hd + 0, w * v.x);
        atomicAdd(hd + 1, w * v.y);
        atomicAdd(hd + 2, w * v.z);
        atomicAdd(hd + 3, w * v.w);
    }
}

// hidden += temp[k] * h
__global__ void axpy_kernel(const float* __restrict__ temp, int k,
                            const float* __restrict__ h, float* __restrict__ out) {
    int idx = blockIdx.x * blockDim.x + threadIdx.x;   // over N*16 float4's
    if (idx < N_NODES * 16) {
        float c  = temp[k];
        float4 v = ((const float4*)h)[idx];
        float4 o = ((float4*)out)[idx];
        o.x += c * v.x; o.y += c * v.y; o.z += c * v.z; o.w += c * v.w;
        ((float4*)out)[idx] = o;
    }
}

// ---------------- driver ----------------

extern "C" void kernel_launch(void* const* d_in, const int* in_sizes, int n_in,
                              void* d_out, int out_size, void* d_ws, size_t ws_size,
                              hipStream_t stream) {
    const float* x    = (const float*)d_in[0];
    const int*   ei   = (const int*)d_in[1];
    const float* W1   = (const float*)d_in[2];
    const float* b1   = (const float*)d_in[3];
    const float* W2   = (const float*)d_in[4];
    const float* b2   = (const float*)d_in[5];
    const float* temp = (const float*)d_in[6];
    float* out = (float*)d_out;

    const int* src = ei;             // edge_index[0]
    const int* dst = ei + N_EDGES;   // edge_index[1]

    // workspace layout (all offsets 16B-aligned)
    char* ws = (char*)d_ws;
    int*      deg  = (int*)ws;                                        // N ints
    float*    dinv = (float*)(ws + (size_t)N_NODES * 4);              // N floats
    float*    hA   = (float*)(ws + (size_t)N_NODES * 8);              // N*64 floats
    float*    hB   = hA + (size_t)N_NODES * NCLASS;                   // N*64 floats
    _Float16* w1t  = (_Float16*)(ws + (size_t)N_NODES * 8
                                   + (size_t)N_NODES * NCLASS * 8);   // 64KB
    _Float16* w2t  = w1t + (size_t)NFEAT * NHID;                      // 8KB

    // gcn_norm
    zero_deg_kernel<<<(N_NODES + 255) / 256, 256, 0, stream>>>(deg);
    deg_kernel<<<(N_EDGES + 255) / 256, 256, 0, stream>>>(dst, deg);
    dinv_kernel<<<(N_NODES + 255) / 256, 256, 0, stream>>>(deg, dinv);

    // MLP (WMMA) -> h0 in hA, out = temp[0]*h0
    prep_weights_kernel<<<(NFEAT * NHID + 255) / 256, 256, 0, stream>>>(W1, W2, w1t, w2t);
    mlp_wmma_kernel<<<(N_NODES + 127) / 128, 256, 0, stream>>>(
        x, w1t, b1, w2t, b2, temp, hA, out);

    // K rounds of propagation
    float* cur = hA;
    float* nxt = hB;
    const int n16 = N_NODES * 16;
    const long long et = (long long)N_EDGES * 16;
    for (int k = 0; k < KPROP; ++k) {
        init_next_kernel<<<(n16 + 255) / 256, 256, 0, stream>>>(cur, dinv, nxt);
        edge_kernel<<<(int)((et + 255) / 256), 256, 0, stream>>>(src, dst, dinv, cur, nxt);
        axpy_kernel<<<(n16 + 255) / 256, 256, 0, stream>>>(temp, k + 1, nxt, out);
        float* t = cur; cur = nxt; nxt = t;
    }
}
